// FactorFNO2dBlock_16406775071264
// MI455X (gfx1250) — matrix operand order repaired and checked by reading, not compiled, not run on hardware
//
#include <hip/hip_runtime.h>
#include <cstdint>

// ---------------- types / helpers ----------------
typedef __attribute__((ext_vector_type(16))) __bf16          bf16x16;
typedef __attribute__((ext_vector_type(16))) unsigned short  u16x16;
typedef __attribute__((ext_vector_type(8)))  unsigned short  u16x8;
typedef __attribute__((ext_vector_type(8)))  float           f32x8;
typedef __attribute__((ext_vector_type(4)))  int             v4i;

__device__ __forceinline__ unsigned short f2bf(float f) {
  unsigned int u = __builtin_bit_cast(unsigned int, f);
  unsigned int r = (u + 0x7FFFu + ((u >> 16) & 1u)) >> 16;
  return (unsigned short)r;
}
__device__ __forceinline__ float bf2f(unsigned short u) {
  unsigned int x = ((unsigned int)u) << 16;
  return __builtin_bit_cast(float, x);
}
__device__ __forceinline__ f32x8 zero8() {
  f32x8 z = {0.f, 0.f, 0.f, 0.f, 0.f, 0.f, 0.f, 0.f};
  return z;
}

// ---- gfx1250 async global->LDS (ASYNCcnt-tracked), guarded fallback ----
#if __has_builtin(__builtin_amdgcn_global_load_async_to_lds_b128) && \
    __has_builtin(__builtin_amdgcn_s_wait_asynccnt)
#define HAVE_ASYNC_LDS 1
#else
#define HAVE_ASYNC_LDS 0
#endif

__device__ __forceinline__ void async_copy16(const unsigned short* g,
                                             unsigned short* l) {
#if HAVE_ASYNC_LDS
  __builtin_amdgcn_global_load_async_to_lds_b128(
      (__attribute__((address_space(1))) v4i*)g,
      (__attribute__((address_space(3))) v4i*)l, 0, 0);
#else
  *(u16x8*)l = *(const u16x8*)g;
#endif
}
__device__ __forceinline__ void async_wait() {
#if HAVE_ASYNC_LDS
  __builtin_amdgcn_s_wait_asynccnt(0);
#endif
}

// D = A(16x32 bf16) x B(32x16 bf16) + C, f32 accum
__device__ __forceinline__ f32x8 wmma_bf16(u16x16 a, u16x16 b, f32x8 c) {
  return __builtin_amdgcn_wmma_f32_16x16x32_bf16(
      false, __builtin_bit_cast(bf16x16, a),
      false, __builtin_bit_cast(bf16x16, b),
      (short)0, c, false, false);
}

// A fragment: tile origin As (row-major [16][K], row stride lda ushorts).
__device__ __forceinline__ u16x16 load_a_frag(const unsigned short* As, int lda) {
  int lane = threadIdx.x & 31;
  int r = lane & 15, half = lane >> 4;
  const unsigned short* p = As + r * lda;
  u16x16 a;
#pragma unroll
  for (int e = 0; e < 8; ++e) a[e] = p[8 * half + e];
#pragma unroll
  for (int e = 0; e < 8; ++e) a[8 + e] = p[16 + 8 * half + e];
  return a;
}
// B fragment from K-major LDS tile Bt[n][k] (row stride ldb ushorts)
__device__ __forceinline__ u16x16 load_b_frag(const unsigned short* Bt, int ldb) {
  int lane = threadIdx.x & 31;
  int r = lane & 15, half = lane >> 4;
  const unsigned short* p = Bt + r * ldb;
  u16x16 b;
#pragma unroll
  for (int e = 0; e < 8; ++e) b[e] = p[8 * half + e];
#pragma unroll
  for (int e = 0; e < 8; ++e) b[8 + e] = p[16 + 8 * half + e];
  return b;
}
// store 16x16 f32 D tile as bf16, unit column stride
__device__ __forceinline__ void store_d_bf16(f32x8 d, unsigned short* out, int ld) {
  int lane = threadIdx.x & 31;
  int r = lane & 15, half = lane >> 4;
#pragma unroll
  for (int v = 0; v < 8; ++v) out[(v + 8 * half) * ld + r] = f2bf(d[v]);
}
// store with explicit row/col strides (elements)
__device__ __forceinline__ void store_d_bf16_rc(f32x8 d, unsigned short* out,
                                                int rs, int cs) {
  int lane = threadIdx.x & 31;
  int r = lane & 15, half = lane >> 4;
#pragma unroll
  for (int v = 0; v < 8; ++v)
    out[(size_t)(v + 8 * half) * rs + (size_t)r * cs] = f2bf(d[v]);
}

#define NB   8
#define NC   64
#define NH   256
#define NW   256
#define NM   32   // retained modes

// ---------------- K0: build DFT / iDFT bases ----------------
// Fr[k][n]=cos(2pi k n/256), Fi=-sin  (rfft)
// iF[h][j]: j<32: s_r*cos ; j>=32: -s_i*sin  (irfft, conj symmetry folded; 1/H incl.)
__global__ void k_build_basis(unsigned short* Fr, unsigned short* Fi,
                              unsigned short* iF) {
  const float TWO_PI = 6.28318530717958647692f;
  int tid = blockIdx.x * blockDim.x + threadIdx.x;
  int nth = gridDim.x * blockDim.x;
  for (int idx = tid; idx < NM * 256; idx += nth) {
    int k = idx >> 8, n = idx & 255;
    float th = TWO_PI * (float)((k * n) & 255) * (1.f / 256.f);
    Fr[idx] = f2bf(__cosf(th));
    Fi[idx] = f2bf(-__sinf(th));
  }
  for (int idx = tid; idx < 256 * 64; idx += nth) {
    int h = idx >> 6, j = idx & 63;
    int k = j & 31;
    bool im = j >= 32;
    float th = TWO_PI * (float)((k * h) & 255) * (1.f / 256.f);
    float v;
    if (im) v = (k == 0) ? 0.f : (-__sinf(th) * (2.f / 256.f));
    else    v = __cosf(th) * ((k == 0) ? (1.f / 256.f) : (2.f / 256.f));
    iF[h * 64 + j] = f2bf(v);
  }
}

// ---------------- K1: DFT along H ----------------
// per block (b,c): X[k,w] = sum_h F[k,h] * x[h,w]   M=32 N=256 K=256
__global__ __launch_bounds__(256) void k_dft_h(
    const float* __restrict__ x, const unsigned short* __restrict__ Fr,
    const unsigned short* __restrict__ Fi,
    unsigned short* __restrict__ Xr, unsigned short* __restrict__ Xi) {
  __shared__ __attribute__((aligned(16))) unsigned short Ar[32 * 40], Ai[32 * 40],
      Bt[256 * 40];
  int bc = blockIdx.x;
  const float* xp = x + (size_t)bc * NH * NW;
  unsigned short* xr = Xr + (size_t)bc * NM * NW;
  unsigned short* xi = Xi + (size_t)bc * NM * NW;
  int tid = threadIdx.x, wave = tid >> 5;
  f32x8 aR[2][2], aI[2][2];
#pragma unroll
  for (int m = 0; m < 2; ++m)
#pragma unroll
    for (int n = 0; n < 2; ++n) { aR[m][n] = zero8(); aI[m][n] = zero8(); }
  for (int kc = 0; kc < 8; ++kc) {
    int h0 = kc * 32;
    if (kc < 7)  // prefetch next chunk of x while computing this one
      __builtin_prefetch(xp + (size_t)(h0 + 32 + (tid >> 3)) * NW + (tid & 7) * 32, 0, 3);
    __syncthreads();
    for (int i = tid; i < 32 * 32; i += 256) {
      int k = i >> 5, kk = i & 31;
      Ar[k * 40 + kk] = Fr[k * 256 + h0 + kk];
      Ai[k * 40 + kk] = Fi[k * 256 + h0 + kk];
    }
    for (int i = tid; i < 32 * 256; i += 256) {
      int kk = i >> 8, w = i & 255;
      Bt[w * 40 + kk] = f2bf(xp[(size_t)(h0 + kk) * NW + w]);
    }
    __syncthreads();
#pragma unroll
    for (int ntl = 0; ntl < 2; ++ntl) {
      int nt = wave * 2 + ntl;
      u16x16 bf = load_b_frag(Bt + nt * 16 * 40, 40);
#pragma unroll
      for (int mt = 0; mt < 2; ++mt) {
        u16x16 ar = load_a_frag(Ar + mt * 16 * 40, 40);
        u16x16 ai = load_a_frag(Ai + mt * 16 * 40, 40);
        aR[mt][ntl] = wmma_bf16(ar, bf, aR[mt][ntl]);
        aI[mt][ntl] = wmma_bf16(ai, bf, aI[mt][ntl]);
      }
    }
  }
#pragma unroll
  for (int ntl = 0; ntl < 2; ++ntl) {
    int nt = wave * 2 + ntl;
#pragma unroll
    for (int mt = 0; mt < 2; ++mt) {
      store_d_bf16(aR[mt][ntl], xr + (mt * 16) * NW + nt * 16, NW);
      store_d_bf16(aI[mt][ntl], xi + (mt * 16) * NW + nt * 16, NW);
    }
  }
}

// ---------------- K4: DFT along W ----------------
// per block (b,c): X[k,h] = sum_w F[k,w] * x[h,w]; store [b][k][c][h]
__global__ __launch_bounds__(256) void k_dft_w(
    const float* __restrict__ x, const unsigned short* __restrict__ Fr,
    const unsigned short* __restrict__ Fi,
    unsigned short* __restrict__ Xr, unsigned short* __restrict__ Xi) {
  __shared__ __attribute__((aligned(16))) unsigned short Ar[32 * 40], Ai[32 * 40],
      Bt[256 * 40];
  int bc = blockIdx.x;
  int b = bc >> 6, c = bc & 63;
  const float* xp = x + (size_t)bc * NH * NW;
  size_t obase = ((size_t)b * NM * NC + c) * NH;  // + k*(NC*NH) + h
  int tid = threadIdx.x, wave = tid >> 5;
  f32x8 aR[2][2], aI[2][2];
#pragma unroll
  for (int m = 0; m < 2; ++m)
#pragma unroll
    for (int n = 0; n < 2; ++n) { aR[m][n] = zero8(); aI[m][n] = zero8(); }
  for (int kc = 0; kc < 8; ++kc) {
    int w0 = kc * 32;
    if (kc < 7)
      __builtin_prefetch(xp + (size_t)tid * NW + w0 + 32, 0, 3);
    __syncthreads();
    for (int i = tid; i < 32 * 32; i += 256) {
      int k = i >> 5, kk = i & 31;
      Ar[k * 40 + kk] = Fr[k * 256 + w0 + kk];
      Ai[k * 40 + kk] = Fi[k * 256 + w0 + kk];
    }
    for (int i = tid; i < 32 * 256; i += 256) {
      int kk = i & 31, h = i >> 5;
      Bt[h * 40 + kk] = f2bf(xp[(size_t)h * NW + w0 + kk]);
    }
    __syncthreads();
#pragma unroll
    for (int ntl = 0; ntl < 2; ++ntl) {
      int nt = wave * 2 + ntl;
      u16x16 bf = load_b_frag(Bt + nt * 16 * 40, 40);
#pragma unroll
      for (int mt = 0; mt < 2; ++mt) {
        u16x16 ar = load_a_frag(Ar + mt * 16 * 40, 40);
        u16x16 ai = load_a_frag(Ai + mt * 16 * 40, 40);
        aR[mt][ntl] = wmma_bf16(ar, bf, aR[mt][ntl]);
        aI[mt][ntl] = wmma_bf16(ai, bf, aI[mt][ntl]);
      }
    }
  }
#pragma unroll
  for (int ntl = 0; ntl < 2; ++ntl) {
    int nt = wave * 2 + ntl;
#pragma unroll
    for (int mt = 0; mt < 2; ++mt) {
      store_d_bf16(aR[mt][ntl], Xr + obase + (size_t)(mt * 16) * (NC * NH) + nt * 16, NC * NH);
      store_d_bf16(aI[mt][ntl], Xi + obase + (size_t)(mt * 16) * (NC * NH) + nt * 16, NC * NH);
    }
  }
}

// ---------------- K2/K5: per-mode complex channel mixing ----------------
// block = (b, mode k). O[o,n] = sum_i W[i,o,k] * X[i,n]  (complex)
// X row i at: b*524288 + k*xKS + i*XR
// O interleaved layout: O[((b*64+o)*256 + n)*64 + (k | 32+k)]
__global__ __launch_bounds__(256) void k_mix(
    const unsigned short* __restrict__ Xr, const unsigned short* __restrict__ Xi,
    const float* __restrict__ Wr, const float* __restrict__ Wi,
    unsigned short* __restrict__ O, int xKS, int XR) {
  __shared__ __attribute__((aligned(16))) unsigned short Ar[64 * 40], Ai[64 * 40],
      An[64 * 40];
  __shared__ __attribute__((aligned(16))) unsigned short Btr[256 * 40], Bti[256 * 40];
  int b = blockIdx.x >> 5, k = blockIdx.x & 31;
  size_t xbase = (size_t)b * 524288 + (size_t)k * xKS;
  size_t obase = (size_t)b * 1048576 + k;  // + o*16384 + n*64 (+32 for imag)
  int tid = threadIdx.x, wave = tid >> 5;
  f32x8 aR[4][2], aI[4][2];
#pragma unroll
  for (int m = 0; m < 4; ++m)
#pragma unroll
    for (int n = 0; n < 2; ++n) { aR[m][n] = zero8(); aI[m][n] = zero8(); }
  for (int kc = 0; kc < 2; ++kc) {
    __syncthreads();
    for (int i = tid; i < 64 * 32; i += 256) {
      int o = i & 63, ii = i >> 6;
      size_t widx = ((size_t)(kc * 32 + ii) * 64 + o) * 32 + k;
      float wr = Wr[widx], wi = Wi[widx];
      Ar[o * 40 + ii] = f2bf(wr);
      Ai[o * 40 + ii] = f2bf(wi);
      An[o * 40 + ii] = f2bf(-wi);
    }
    for (int i = tid; i < 32 * 256; i += 256) {
      int ii = i >> 8, n = i & 255;
      size_t s = xbase + (size_t)(kc * 32 + ii) * XR + n;
      Btr[n * 40 + ii] = Xr[s];
      Bti[n * 40 + ii] = Xi[s];
    }
    __syncthreads();
#pragma unroll
    for (int ntl = 0; ntl < 2; ++ntl) {
      int nt = wave * 2 + ntl;
      u16x16 br = load_b_frag(Btr + nt * 16 * 40, 40);
      u16x16 bi = load_b_frag(Bti + nt * 16 * 40, 40);
#pragma unroll
      for (int mt = 0; mt < 4; ++mt) {
        u16x16 ar = load_a_frag(Ar + mt * 16 * 40, 40);
        u16x16 ai = load_a_frag(Ai + mt * 16 * 40, 40);
        u16x16 an = load_a_frag(An + mt * 16 * 40, 40);
        aR[mt][ntl] = wmma_bf16(ar, br, aR[mt][ntl]);
        aR[mt][ntl] = wmma_bf16(an, bi, aR[mt][ntl]);
        aI[mt][ntl] = wmma_bf16(ar, bi, aI[mt][ntl]);
        aI[mt][ntl] = wmma_bf16(ai, br, aI[mt][ntl]);
      }
    }
  }
#pragma unroll
  for (int ntl = 0; ntl < 2; ++ntl) {
    int nt = wave * 2 + ntl;
#pragma unroll
    for (int mt = 0; mt < 4; ++mt) {
      unsigned short* po = O + obase + (size_t)(mt * 16) * 16384 + (size_t)(nt * 16) * 64;
      store_d_bf16_rc(aR[mt][ntl], po, 16384, 64);
      store_d_bf16_rc(aI[mt][ntl], po + 32, 16384, 64);
    }
  }
}

// ---------------- K3: iDFT along H (writes f) ----------------
// per block (b,o): y[h,w] = sum_j iF[h,j] * O[w][j]   K=64, O rows contiguous
__global__ __launch_bounds__(256) void k_idft_h(
    const unsigned short* __restrict__ O, const unsigned short* __restrict__ iF,
    unsigned short* __restrict__ f) {
  __shared__ __attribute__((aligned(16))) unsigned short As[128 * 72], Bt[256 * 72];
  int bo = blockIdx.x;
  const unsigned short* op = O + (size_t)bo * 16384;  // [n=256][64]
  unsigned short* fp = f + (size_t)bo * NH * NW;
  int tid = threadIdx.x, wave = tid >> 5;
  // stage B: 256 rows x 128B, contiguous -> async copy
  for (int i = tid; i < 256 * 8; i += 256) {
    int row = i >> 3, ch = i & 7;
    async_copy16(op + row * 64 + ch * 8, Bt + row * 72 + ch * 8);
  }
  for (int phase = 0; phase < 2; ++phase) {
    __syncthreads();
    for (int i = tid; i < 128 * 8; i += 256) {
      int r = i >> 3, ch = i & 7;
      async_copy16(iF + (phase * 128 + r) * 64 + ch * 8, As + r * 72 + ch * 8);
    }
    async_wait();
    __syncthreads();
    int mt = wave;
    for (int nt = 0; nt < 16; ++nt) {
      f32x8 acc = zero8();
#pragma unroll
      for (int kc = 0; kc < 2; ++kc) {
        u16x16 a = load_a_frag(As + (mt * 16) * 72 + kc * 32, 72);
        u16x16 bf = load_b_frag(Bt + (nt * 16) * 72 + kc * 32, 72);
        acc = wmma_bf16(a, bf, acc);
      }
      store_d_bf16(acc, fp + (size_t)(phase * 128 + mt * 16) * NW + nt * 16, NW);
    }
  }
}

// ---------------- K6: iDFT along W (accumulates into f) ----------------
// per block (b,o): y[h,w] = sum_j O[h][j] * iF[w][j]
__global__ __launch_bounds__(256) void k_idft_w_acc(
    const unsigned short* __restrict__ O, const unsigned short* __restrict__ iF,
    unsigned short* __restrict__ f) {
  __shared__ __attribute__((aligned(16))) unsigned short As[128 * 72], Bt[256 * 72];
  int bo = blockIdx.x;
  const unsigned short* op = O + (size_t)bo * 16384;  // [n=256][64]
  unsigned short* fp = f + (size_t)bo * NH * NW;
  int tid = threadIdx.x, wave = tid >> 5;
  int lane = tid & 31, rr = lane & 15, half = lane >> 4;
  for (int i = tid; i < 256 * 8; i += 256) {
    int row = i >> 3, ch = i & 7;
    async_copy16(iF + row * 64 + ch * 8, Bt + row * 72 + ch * 8);
  }
  for (int phase = 0; phase < 2; ++phase) {
    __syncthreads();
    for (int i = tid; i < 128 * 8; i += 256) {
      int r = i >> 3, ch = i & 7;
      async_copy16(op + (phase * 128 + r) * 64 + ch * 8, As + r * 72 + ch * 8);
    }
    async_wait();
    __syncthreads();
    int mt = wave;
    for (int nt = 0; nt < 16; ++nt) {
      f32x8 acc = zero8();
#pragma unroll
      for (int kc = 0; kc < 2; ++kc) {
        u16x16 a = load_a_frag(As + (mt * 16) * 72 + kc * 32, 72);
        u16x16 bf = load_b_frag(Bt + (nt * 16) * 72 + kc * 32, 72);
        acc = wmma_bf16(a, bf, acc);
      }
      int row0 = phase * 128 + mt * 16;
#pragma unroll
      for (int v = 0; v < 8; ++v) {
        size_t idx = (size_t)(row0 + v + 8 * half) * NW + nt * 16 + rr;
        fp[idx] = f2bf(bf2f(fp[idx]) + acc[v]);
      }
    }
  }
}

// ---------------- K7: FFN (GEMM->ReLU->LN->GEMM) + residual ----------------
__global__ __launch_bounds__(128) void k_ffn(
    const unsigned short* __restrict__ f, const float* __restrict__ x,
    const float* __restrict__ fc1w, const float* __restrict__ fc1b,
    const float* __restrict__ lng, const float* __restrict__ lnb,
    const float* __restrict__ fc2w, const float* __restrict__ fc2b,
    float* __restrict__ out) {
  __shared__ __attribute__((aligned(16))) unsigned short As[64 * 72];
  __shared__ __attribute__((aligned(16))) unsigned short Bt1[128 * 72];
  __shared__ __attribute__((aligned(16))) unsigned short Bt2[64 * 136];
  __shared__ __attribute__((aligned(16))) unsigned short At1[64 * 136];
  __shared__ float b1s[128], gs[128], bs[128], b2s[64];
  int tid = threadIdx.x, wave = tid >> 5;
  int lane = tid & 31, rr = lane & 15, half = lane >> 4;
  int p0 = blockIdx.x * 64;
  int b = p0 >> 16;
  int hw0 = p0 & 65535;
  int h = hw0 >> 8, w0 = hw0 & 255;
  for (int i = tid; i < 64 * 64; i += 128) {
    int c = i >> 6, p = i & 63;
    As[p * 72 + c] = f[(((size_t)b * NC + c) * NH + h) * NW + w0 + p];
  }
  for (int i = tid; i < 64 * 128; i += 128) {
    int k = i >> 7, n = i & 127;
    Bt1[n * 72 + k] = f2bf(fc1w[k * 128 + n]);
  }
  for (int i = tid; i < 128 * 64; i += 128) {
    int k = i >> 6, n = i & 63;
    Bt2[n * 136 + k] = f2bf(fc2w[k * 64 + n]);
  }
  if (tid < 128) { b1s[tid] = fc1b[tid]; gs[tid] = lng[tid]; bs[tid] = lnb[tid]; }
  if (tid < 64) b2s[tid] = fc2b[tid];
  __syncthreads();
  int mt = wave;
  for (int nt = 0; nt < 8; ++nt) {
    f32x8 acc = zero8();
#pragma unroll
    for (int kc = 0; kc < 2; ++kc) {
      u16x16 a = load_a_frag(As + (mt * 16) * 72 + kc * 32, 72);
      u16x16 bf = load_b_frag(Bt1 + (nt * 16) * 72 + kc * 32, 72);
      acc = wmma_bf16(a, bf, acc);
    }
    float bias = b1s[nt * 16 + rr];
#pragma unroll
    for (int v = 0; v < 8; ++v) {
      float t = acc[v] + bias;
      t = t > 0.f ? t : 0.f;
      At1[(mt * 16 + v + 8 * half) * 136 + nt * 16 + rr] = f2bf(t);
    }
  }
  __syncthreads();
  if (tid < 64) {  // LayerNorm over 128 features, one row per thread
    float s = 0.f, s2 = 0.f;
    for (int n = 0; n < 128; ++n) {
      float t = bf2f(At1[tid * 136 + n]);
      s += t; s2 += t * t;
    }
    float mu = s * (1.f / 128.f);
    float var = s2 * (1.f / 128.f) - mu * mu;
    float inv = rsqrtf(var + 1e-5f);
    for (int n = 0; n < 128; ++n) {
      float t = bf2f(At1[tid * 136 + n]);
      At1[tid * 136 + n] = f2bf((t - mu) * inv * gs[n] + bs[n]);
    }
  }
  __syncthreads();
  for (int nt = 0; nt < 4; ++nt) {
    f32x8 acc = zero8();
#pragma unroll
    for (int kc = 0; kc < 4; ++kc) {
      u16x16 a = load_a_frag(At1 + (mt * 16) * 136 + kc * 32, 136);
      u16x16 bf = load_b_frag(Bt2 + (nt * 16) * 136 + kc * 32, 136);
      acc = wmma_bf16(a, bf, acc);
    }
    int ch = nt * 16 + rr;
    float bias = b2s[ch];
#pragma unroll
    for (int v = 0; v < 8; ++v) {
      int p = mt * 16 + v + 8 * half;
      size_t oidx = (((size_t)b * NC + ch) * NH + h) * NW + w0 + p;
      out[oidx] = acc[v] + bias + x[oidx];
    }
  }
}

// ---------------- host launcher ----------------
extern "C" void kernel_launch(void* const* d_in, const int* in_sizes, int n_in,
                              void* d_out, int out_size, void* d_ws, size_t ws_size,
                              hipStream_t stream) {
  const float* x     = (const float*)d_in[0];
  const float* wh_r  = (const float*)d_in[1];
  const float* wh_i  = (const float*)d_in[2];
  const float* ww_r  = (const float*)d_in[3];
  const float* ww_i  = (const float*)d_in[4];
  const float* fc1w  = (const float*)d_in[5];
  const float* fc1b  = (const float*)d_in[6];
  const float* lng   = (const float*)d_in[7];
  const float* lnb   = (const float*)d_in[8];
  const float* fc2w  = (const float*)d_in[9];
  const float* fc2b  = (const float*)d_in[10];
  float* out = (float*)d_out;

  // workspace layout (all bf16 as ushort)
  unsigned short* ws = (unsigned short*)d_ws;
  const size_t nX = (size_t)NB * NC * NM * NW;          // 4,194,304
  unsigned short* Xr   = ws;                            // reused by W path
  unsigned short* Xi   = Xr + nX;
  unsigned short* Obuf = Xi + nX;                       // [B*C][256][64] interleaved r/i
  unsigned short* fbuf = Obuf + (size_t)NB * NC * 256 * 64;  // [B][C][H][W] bf16
  unsigned short* Fr   = fbuf + (size_t)NB * NC * NH * NW;
  unsigned short* Fi   = Fr + NM * 256;
  unsigned short* iF   = Fi + NM * 256;                 // [256][64]

  k_build_basis<<<64, 256, 0, stream>>>(Fr, Fi, iF);

  // --- H path ---
  k_dft_h<<<NB * NC, 256, 0, stream>>>(x, Fr, Fi, Xr, Xi);
  k_mix<<<NB * NM, 256, 0, stream>>>(Xr, Xi, wh_r, wh_i, Obuf,
                                     /*xKS=*/256, /*XR=*/NM * NW);
  k_idft_h<<<NB * NC, 256, 0, stream>>>(Obuf, iF, fbuf);

  // --- W path (reuses X/O buffers; stream ordering serializes) ---
  k_dft_w<<<NB * NC, 256, 0, stream>>>(x, Fr, Fi, Xr, Xi);
  k_mix<<<NB * NM, 256, 0, stream>>>(Xr, Xi, ww_r, ww_i, Obuf,
                                     /*xKS=*/NC * NH, /*XR=*/NH);
  k_idft_w_acc<<<NB * NC, 256, 0, stream>>>(Obuf, iF, fbuf);

  // --- FFN + residual ---
  k_ffn<<<(NB * NH * NW) / 64, 128, 0, stream>>>(fbuf, x, fc1w, fc1b, lng, lnb,
                                                 fc2w, fc2b, out);
  (void)in_sizes; (void)n_in; (void)out_size; (void)ws_size;
}
